// LSTMClassifier_45698452029522
// MI455X (gfx1250) — matrix operand order repaired and checked by reading, not compiled, not run on hardware
//
#include <hip/hip_runtime.h>

typedef __attribute__((ext_vector_type(16))) _Float16 v16h;
typedef __attribute__((ext_vector_type(8)))  float    v8f;

#define Bdim  512
#define Tdim  2048
#define Hdim  64
#define NCLSd 10
#define NG    256     // 4*H gate columns
#define CHUNK 128     // x staging chunk (timesteps)
#define HSTR  72      // padded f16 row stride for hidden state (bank spread)
#define GSTR  260     // padded f32 row stride for gates buffer

// Fast reciprocal: single v_rcp_f32
__device__ __forceinline__ float frcp(float x)   { return __builtin_amdgcn_rcpf(x); }
__device__ __forceinline__ float fsig(float x)   { return frcp(1.0f + __expf(-x)); }
// tanh(x) = 1 - 2/(1 + e^{2x})
__device__ __forceinline__ float ftanhf(float x) { return 1.0f - 2.0f * frcp(1.0f + __expf(2.0f * x)); }

// A fragment (16x32 f16, M x K): lane m = lane&15; low lanes K=[0..7,16..23], high lanes +8.
__device__ __forceinline__ v16h load_afrag(const _Float16* Hs, int kb, int lane) {
    int m  = lane & 15;
    int k0 = kb * 32 + ((lane & 16) ? 8 : 0);
    const _Float16* p = Hs + m * HSTR + k0;
    v16h r;
#pragma unroll
    for (int i = 0; i < 8; i++) r[i] = p[i];
#pragma unroll
    for (int i = 0; i < 8; i++) r[8 + i] = p[16 + i];
    return r;
}

// B fragment (32x16 f16, K x N) from W[256][64] (row n holds K contiguous):
// n = nt*16 + (lane&15); low lanes K=[0..7,16..23], high lanes +8.
__device__ __forceinline__ v16h load_bfrag(const float* W, int nt, int kb, int lane) {
    int n  = nt * 16 + (lane & 15);
    int k0 = kb * 32 + ((lane & 16) ? 8 : 0);
    const float* p = W + n * Hdim + k0;
    v16h r;
#pragma unroll
    for (int i = 0; i < 8; i++) r[i] = (_Float16)p[i];
#pragma unroll
    for (int i = 0; i < 8; i++) r[8 + i] = (_Float16)p[16 + i];
    return r;
}

// C/D f32 16x16: reg r -> row r + 8*(lane>=16), col lane&15.
__device__ __forceinline__ void store_cfrag(float* G, int nt, int lane, v8f acc) {
    int n     = nt * 16 + (lane & 15);
    int mbase = (lane & 16) ? 8 : 0;
#pragma unroll
    for (int r = 0; r < 8; r++) G[(mbase + r) * GSTR + n] = acc[r];
}

__global__ __launch_bounds__(256)
void lstm_cls_fused_kernel(const float* __restrict__ x,
                           const float* __restrict__ w_ih0, const float* __restrict__ w_hh0,
                           const float* __restrict__ b_ih0, const float* __restrict__ b_hh0,
                           const float* __restrict__ w_ih1, const float* __restrict__ w_hh1,
                           const float* __restrict__ b_ih1, const float* __restrict__ b_hh1,
                           const float* __restrict__ w1, const float* __restrict__ b1,
                           const float* __restrict__ w2, const float* __restrict__ b2,
                           const float* __restrict__ w3, const float* __restrict__ b3,
                           float* __restrict__ out) {
    __shared__ _Float16 sh_h0[16 * HSTR];   // layer0 hidden (f16, WMMA A source)
    __shared__ _Float16 sh_h1[16 * HSTR];   // layer1 hidden (f16)
    __shared__ float sh_h1f[16 * Hdim];     // layer1 hidden (f32, for classifier; written once)
    __shared__ float sh_g[16 * GSTR];       // gate pre-activations / head scratch
    __shared__ float sh_x[16 * CHUNK];      // staged input chunk

    const int tid  = threadIdx.x;
    const int lane = tid & 31;
    const int wv   = tid >> 5;              // wave 0..7; owns N-tiles wv and wv+8
    const int b0   = blockIdx.x * 16;

    // ---- init hidden state ----
    for (int i = tid; i < 16 * HSTR; i += 256) { sh_h0[i] = (_Float16)0.f; sh_h1[i] = (_Float16)0.f; }

    // ---- per-thread elementwise constants (j is the SAME for all 4 owned elements) ----
    const int je = tid & 63;                // gate column within H
    const int m0 = tid >> 6;                // owned rows: m0 + 4r, r=0..3
    float wih0g[4], b0g[4], b1g[4];
#pragma unroll
    for (int g = 0; g < 4; g++) {
        int n = 64 * g + je;
        wih0g[g] = w_ih0[n];
        b0g[g]   = b_ih0[n] + b_hh0[n];
        b1g[g]   = b_ih1[n] + b_hh1[n];
    }
    // cell states + final hidden live in registers (exclusively owned per thread)
    float c0r[4] = {0.f, 0.f, 0.f, 0.f};
    float c1r[4] = {0.f, 0.f, 0.f, 0.f};
    float h1r[4] = {0.f, 0.f, 0.f, 0.f};

    // ---- preload all weight B-fragments into VGPRs (loop-invariant) ----
    v16h Bhh0[2][2], Bih1[2][2], Bhh1[2][2];
#pragma unroll
    for (int j = 0; j < 2; j++) {
        int nt = wv + 8 * j;
#pragma unroll
        for (int kb = 0; kb < 2; kb++) {
            Bhh0[j][kb] = load_bfrag(w_hh0, nt, kb, lane);
            Bih1[j][kb] = load_bfrag(w_ih1, nt, kb, lane);
            Bhh1[j][kb] = load_bfrag(w_hh1, nt, kb, lane);
        }
    }
    __syncthreads();

    for (int tc = 0; tc < Tdim; tc += CHUNK) {
        // stage x[b0..b0+15][tc..tc+127] coalesced
        {
            int r  = tid >> 4;
            int c0 = (tid & 15) * 8;
            const float* src = x + (size_t)(b0 + r) * Tdim + tc + c0;
#pragma unroll
            for (int i = 0; i < 8; i++) sh_x[r * CHUNK + c0 + i] = src[i];
        }
        __syncthreads();

        for (int tt = 0; tt < CHUNK; ++tt) {
            // ==== phase 1: layer0 GEMM (h0 @ w_hh0^T) -> sh_g,
            //               layer1 recurrent GEMM (h1 @ w_hh1^T) -> registers ====
            v8f acc1[2];
            {
                v16h a0 = load_afrag(sh_h0, 0, lane);
                v16h a1 = load_afrag(sh_h0, 1, lane);
                v16h r0 = load_afrag(sh_h1, 0, lane);
                v16h r1 = load_afrag(sh_h1, 1, lane);
#pragma unroll
                for (int j = 0; j < 2; j++) {
                    v8f acc = {};
                    acc = __builtin_amdgcn_wmma_f32_16x16x32_f16(false, a0, false, Bhh0[j][0], (short)0, acc, false, false);
                    acc = __builtin_amdgcn_wmma_f32_16x16x32_f16(false, a1, false, Bhh0[j][1], (short)0, acc, false, false);
                    store_cfrag(sh_g, wv + 8 * j, lane, acc);
                    v8f ac = {};
                    ac = __builtin_amdgcn_wmma_f32_16x16x32_f16(false, r0, false, Bhh1[j][0], (short)0, ac, false, false);
                    ac = __builtin_amdgcn_wmma_f32_16x16x32_f16(false, r1, false, Bhh1[j][1], (short)0, ac, false, false);
                    acc1[j] = ac;
                }
            }
            __syncthreads();
            // ==== phase 2: elementwise layer 0 (x*w_ih0 + biases; gate order i,f,g,o) ====
#pragma unroll
            for (int r = 0; r < 4; r++) {
                int m = m0 + 4 * r;
                float xt = sh_x[m * CHUNK + tt];
                float gi = sh_g[m * GSTR + je]       + xt * wih0g[0] + b0g[0];
                float gf = sh_g[m * GSTR + 64 + je]  + xt * wih0g[1] + b0g[1];
                float gg = sh_g[m * GSTR + 128 + je] + xt * wih0g[2] + b0g[2];
                float go = sh_g[m * GSTR + 192 + je] + xt * wih0g[3] + b0g[3];
                float iv = fsig(gi), fv = fsig(gf), gv = ftanhf(gg), ov = fsig(go);
                float c = fv * c0r[r] + iv * gv;
                c0r[r] = c;
                sh_h0[m * HSTR + je] = (_Float16)(ov * ftanhf(c));
            }
            __syncthreads();
            // ==== phase 3: layer1 input GEMM (h0_cur @ w_ih1^T), accumulate onto acc1 ====
            {
                v16h x0 = load_afrag(sh_h0, 0, lane);
                v16h x1 = load_afrag(sh_h0, 1, lane);
#pragma unroll
                for (int j = 0; j < 2; j++) {
                    v8f ac = acc1[j];
                    ac = __builtin_amdgcn_wmma_f32_16x16x32_f16(false, x0, false, Bih1[j][0], (short)0, ac, false, false);
                    ac = __builtin_amdgcn_wmma_f32_16x16x32_f16(false, x1, false, Bih1[j][1], (short)0, ac, false, false);
                    store_cfrag(sh_g, wv + 8 * j, lane, ac);
                }
            }
            __syncthreads();
            // ==== phase 4: elementwise layer 1 ====
#pragma unroll
            for (int r = 0; r < 4; r++) {
                int m = m0 + 4 * r;
                float gi = fsig  (sh_g[m * GSTR + je]       + b1g[0]);
                float gf = fsig  (sh_g[m * GSTR + 64 + je]  + b1g[1]);
                float gg = ftanhf(sh_g[m * GSTR + 128 + je] + b1g[2]);
                float go = fsig  (sh_g[m * GSTR + 192 + je] + b1g[3]);
                float c = gf * c1r[r] + gi * gg;
                c1r[r] = c;
                float h = go * ftanhf(c);
                h1r[r] = h;
                sh_h1[m * HSTR + je] = (_Float16)h;
            }
            __syncthreads();
        }
    }

    // publish final h1 (f32) for the classifier head
#pragma unroll
    for (int r = 0; r < 4; r++) sh_h1f[(m0 + 4 * r) * Hdim + je] = h1r[r];
    __syncthreads();

    // ======== classifier head on final h1 (16x64) ========
    // y1 = relu(h @ w1^T + b1) : 16x64 -> sh_g cols [0,64)
#pragma unroll
    for (int r = 0; r < 4; r++) {
        int m = m0 + 4 * r;
        float s = b1[je];
        for (int k = 0; k < 64; k++) s += sh_h1f[m * Hdim + k] * w1[je * 64 + k];
        sh_g[m * GSTR + je] = fmaxf(s, 0.f);
    }
    __syncthreads();
    // y2 = relu(y1 @ w2^T + b2) : 16x32 -> sh_g cols [64,96)
#pragma unroll
    for (int r = 0; r < 2; r++) {
        int e = tid + 256 * r;
        int m = e >> 5, j = e & 31;
        float s = b2[j];
        for (int k = 0; k < 64; k++) s += sh_g[m * GSTR + k] * w2[j * 64 + k];
        sh_g[m * GSTR + 64 + j] = fmaxf(s, 0.f);
    }
    __syncthreads();
    // out = y2 @ w3^T + b3 : 16x10
    if (tid < 16 * NCLSd) {
        int m = tid / NCLSd, j = tid % NCLSd;
        float s = b3[j];
        for (int k = 0; k < 32; k++) s += sh_g[m * GSTR + 64 + k] * w3[j * 32 + k];
        out[(size_t)(b0 + m) * NCLSd + j] = s;
    }
}

extern "C" void kernel_launch(void* const* d_in, const int* in_sizes, int n_in,
                              void* d_out, int out_size, void* d_ws, size_t ws_size,
                              hipStream_t stream) {
    (void)in_sizes; (void)n_in; (void)d_ws; (void)ws_size; (void)out_size;
    const float* x     = (const float*)d_in[0];
    const float* w_ih0 = (const float*)d_in[1];
    const float* w_hh0 = (const float*)d_in[2];
    const float* b_ih0 = (const float*)d_in[3];
    const float* b_hh0 = (const float*)d_in[4];
    const float* w_ih1 = (const float*)d_in[5];
    const float* w_hh1 = (const float*)d_in[6];
    const float* b_ih1 = (const float*)d_in[7];
    const float* b_hh1 = (const float*)d_in[8];
    const float* w1    = (const float*)d_in[9];
    const float* b1    = (const float*)d_in[10];
    const float* w2    = (const float*)d_in[11];
    const float* b2    = (const float*)d_in[12];
    const float* w3    = (const float*)d_in[13];
    const float* b3    = (const float*)d_in[14];
    float* out = (float*)d_out;

    dim3 grid(Bdim / 16);   // 32 workgroups, one 16-row batch tile each
    dim3 block(256);        // 8 waves (wave32)
    lstm_cls_fused_kernel<<<grid, block, 0, stream>>>(
        x, w_ih0, w_hh0, b_ih0, b_hh0, w_ih1, w_hh1, b_ih1, b_hh1,
        w1, b1, w2, b2, w3, b3, out);
}
